// VNGNN_59004260712941
// MI455X (gfx1250) — matrix operand order repaired and checked by reading, not compiled, not run on hardware
//
#include <hip/hip_runtime.h>

// ---------------------------------------------------------------------------
// Problem constants (from reference): B=524288 nodes, C=256 clusters, D=128
// ---------------------------------------------------------------------------
#define NB 524288
#define NC 256
#define ND 128

typedef __attribute__((ext_vector_type(16))) __bf16 v16bf;
typedef __attribute__((ext_vector_type(8)))  __bf16 v8bf;
typedef __attribute__((ext_vector_type(8)))  float  v8f;

// --- WMMA fragment builders (CDNA5 16x16x32 bf16 layouts, ISA 7.12.2) ------
// A matrix 16x32, row-major source row (128 bf16 per row in LDS):
//   lane holds row M=lane%16;  a[j] = A[M][k0 + (j<8?j:j+8) + (lane>=16?8:0)]
__device__ inline v16bf frag_a_bf(const __bf16* __restrict__ row, int k0, int lane) {
    const int off = (lane >= 16) ? 8 : 0;
    v8bf lo = *(const v8bf*)(row + k0 + off);
    v8bf hi = *(const v8bf*)(row + k0 + off + 16);
    v16bf a;
#pragma unroll
    for (int i = 0; i < 8; ++i) { a[i] = lo[i]; a[i + 8] = hi[i]; }
    return a;
}
// A fragment built from an f32 LDS tile (convert on the fly)
__device__ inline v16bf frag_a_f32(const float* __restrict__ row, int k0, int lane) {
    const int off = (lane >= 16) ? 8 : 0;
    v16bf a;
#pragma unroll
    for (int i = 0; i < 8; ++i) {
        a[i]     = (__bf16)row[k0 + off + i];
        a[i + 8] = (__bf16)row[k0 + off + 16 + i];
    }
    return a;
}
// B matrix 32x16 (Bmat[k][n] = W[n][k], W row-major in LDS):
//   lane holds col N=lane%16;  b[j] = W[N][k0 + (j<8?j:j+8) + (lane>=16?8:0)]
__device__ inline v16bf frag_b_bf(const __bf16* __restrict__ wrow, int k0, int lane) {
    const int off = (lane >= 16) ? 8 : 0;
    v8bf lo = *(const v8bf*)(wrow + k0 + off);
    v8bf hi = *(const v8bf*)(wrow + k0 + off + 16);
    v16bf b;
#pragma unroll
    for (int i = 0; i < 8; ++i) { b[i] = lo[i]; b[i + 8] = hi[i]; }
    return b;
}

__device__ inline float sigmoidf_(float x) { return 1.0f / (1.0f + __expf(-x)); }

// monotonic float<->uint encoding for atomic max over signed floats
__device__ inline unsigned enc_f32(float f) {
    unsigned b = __float_as_uint(f);
    return (b & 0x80000000u) ? ~b : (b | 0x80000000u);
}
__device__ inline float dec_max(unsigned u) {
    if (u == 0u) return 0.0f; // untouched (empty cluster) -> reference maps to 0
    unsigned b = (u & 0x80000000u) ? (u ^ 0x80000000u) : ~u;
    return __uint_as_float(b);
}

// ---------------------------------------------------------------------------
// Kernel 1: C-side precompute.   g_trans = tanh(g@Fs^T + Fs_b)
//   q = g@Wq^T ;  u1 = g_trans@G1^T + G1_b + H1_b
// ---------------------------------------------------------------------------
__global__ __launch_bounds__(ND) void k_cside_pre(
    const float* __restrict__ g, const float* __restrict__ Fs_w,
    const float* __restrict__ Fs_b, const float* __restrict__ Wq,
    const float* __restrict__ G1_w, const float* __restrict__ G1_b,
    const float* __restrict__ H1_b,
    float* __restrict__ g_trans, float* __restrict__ q, float* __restrict__ u1)
{
    __shared__ float grow[ND], gtrow[ND];
    const int c = blockIdx.x, d = threadIdx.x;
    grow[d] = g[c * ND + d];
    __syncthreads();
    float s1 = Fs_b[d], s2 = 0.0f;
    const float* fr = Fs_w + d * ND;
    const float* qr = Wq + d * ND;
#pragma unroll 4
    for (int k = 0; k < ND; ++k) { s1 += grow[k] * fr[k]; s2 += grow[k] * qr[k]; }
    float gt = tanhf(s1);
    gtrow[d] = gt;
    g_trans[c * ND + d] = gt;
    q[c * ND + d] = s2;
    __syncthreads();
    float s3 = G1_b[d] + H1_b[d];
    const float* gr = G1_w + d * ND;
#pragma unroll 4
    for (int k = 0; k < ND; ++k) s3 += gtrow[k] * gr[k];
    u1[c * ND + d] = s3;
}

// ---------------------------------------------------------------------------
// Kernel 2: pass 1 over B. 64 rows/block, 8 waves.
//   k = h@Wk^T (WMMA bf16), score = tanh(k + q[cid] + bias) . Ws_w + Ws_b
//   atomic segment-max of score per cluster.
// LDS: hbf 64x128 bf16 (16KB) | wkbf 128x128 bf16 (32KB) | kf 64x128 f32 (32KB)
// ---------------------------------------------------------------------------
__global__ __launch_bounds__(256) void k_pass1(
    const float* __restrict__ h, const float* __restrict__ Wk,
    const int* __restrict__ cid, const float* __restrict__ q,
    const float* __restrict__ bias, const float* __restrict__ Ws_w,
    const float* __restrict__ Ws_b,
    float* __restrict__ score, unsigned* __restrict__ m_enc)
{
    extern __shared__ char smem[];
    __bf16* hbf  = (__bf16*)smem;                               // 64*128 bf16
    __bf16* wkbf = (__bf16*)(smem + 64 * ND * 2);               // 128*128 bf16
    float*  kf   = (float*)(smem + 64 * ND * 2 + ND * ND * 2);  // 64*128 f32

    const int tid  = threadIdx.x;
    const int row0 = blockIdx.x * 64;

    // stage h tile (f32 -> bf16) and Wk (f32 -> bf16), both row-major
    const float4* hsrc = (const float4*)(h + (size_t)row0 * ND);
    for (int i = tid; i < 64 * ND / 4; i += 256) {
        float4 v = hsrc[i];
        int b = i * 4;
        hbf[b] = (__bf16)v.x; hbf[b + 1] = (__bf16)v.y;
        hbf[b + 2] = (__bf16)v.z; hbf[b + 3] = (__bf16)v.w;
    }
    const float4* wsrc = (const float4*)Wk;
    for (int i = tid; i < ND * ND / 4; i += 256) {
        float4 v = wsrc[i];
        int b = i * 4;
        wkbf[b] = (__bf16)v.x; wkbf[b + 1] = (__bf16)v.y;
        wkbf[b + 2] = (__bf16)v.z; wkbf[b + 3] = (__bf16)v.w;
    }
    __syncthreads();

    // 8 waves: wave = 2*rowTile + colHalf; each wave owns 16 rows x 64 cols
    const int wave = tid >> 5, lane = tid & 31;
    const int rt = wave >> 1;   // row tile 0..3
    const int ch = wave & 1;    // column half 0..1
    v8f acc[4] = {v8f{}, v8f{}, v8f{}, v8f{}};
    const __bf16* arow = hbf + (rt * 16 + (lane & 15)) * ND;
#pragma unroll
    for (int kc = 0; kc < 4; ++kc) {
        const int k0 = kc * 32;
        v16bf a = frag_a_bf(arow, k0, lane);
#pragma unroll
        for (int c = 0; c < 4; ++c) {
            const int n = ((ch * 4 + c) * 16) + (lane & 15);
            v16bf b = frag_b_bf(wkbf + n * ND, k0, lane);
            acc[c] = __builtin_amdgcn_wmma_f32_16x16x32_bf16(
                false, a, false, b, (short)0, acc[c], false, false);
        }
    }
    // write k tiles to LDS (C/D layout: M = v + (lane>=16?8:0), N = lane%16)
    const int Mb = rt * 16 + ((lane >= 16) ? 8 : 0);
#pragma unroll
    for (int c = 0; c < 4; ++c) {
        const int N = (ch * 4 + c) * 16 + (lane & 15);
#pragma unroll
        for (int v = 0; v < 8; ++v) kf[(Mb + v) * ND + N] = acc[c][v];
    }
    __syncthreads();

    // score: 4 threads per row, each covers a 32-wide quarter of D
    const int r  = tid >> 2;
    const int qd = tid & 3;
    const int gi = row0 + r;
    const int cc = cid[gi];
    const float* qrow = q + cc * ND;
    float p = 0.0f;
    const int nbeg = qd * 32;
#pragma unroll 4
    for (int n = nbeg; n < nbeg + 32; ++n)
        p += tanhf(kf[r * ND + n] + qrow[n] + bias[n]) * Ws_w[n];
    p += __shfl_xor(p, 1, 32);
    p += __shfl_xor(p, 2, 32);
    if (qd == 0) {
        float s = p + Ws_b[0];
        score[gi] = s;
        atomicMax(&m_enc[cc], enc_f32(s));
    }
}

// ---------------------------------------------------------------------------
// Kernel 3: pass 2 over B. 1024 rows/block (16 chunks of 64), 8 waves.
//   e = exp(score - m[cid]); LDS-accumulate denom[c] and ctx[c][:] += e*h
//   a1 = h@H1^T (WMMA); z1 = sigmoid(a1 + u1[cid]); h_merged written to out.
// LDS: ctab 256x128 f32 (128KB) | dtab 256 f32 | h1bf 128x128 bf16 (32KB)
//      | ht 64x128 f32 (32KB) | erow 64 f32 | crow 64 i32   (~194KB total)
// ---------------------------------------------------------------------------
__global__ __launch_bounds__(256) void k_pass2(
    const float* __restrict__ h, const int* __restrict__ cid,
    const float* __restrict__ H1_w, const float* __restrict__ u1,
    const float* __restrict__ g_trans, const float* __restrict__ score,
    const unsigned* __restrict__ m_enc,
    float* __restrict__ denom, float* __restrict__ ctxe, float* __restrict__ out)
{
    extern __shared__ char smem[];
    float*  ctab = (float*)smem;                                  // NC*ND f32
    float*  dtab = (float*)(smem + NC * ND * 4);                  // NC f32
    __bf16* h1bf = (__bf16*)(smem + NC * ND * 4 + NC * 4);        // ND*ND bf16
    float*  ht   = (float*)(smem + NC * ND * 4 + NC * 4 + ND * ND * 2); // 64*ND f32
    float*  erow = (float*)((char*)ht + 64 * ND * 4);             // 64 f32
    int*    crow = (int*)((char*)erow + 64 * 4);                  // 64 i32

    const int tid = threadIdx.x;
    const int wave = tid >> 5, lane = tid & 31;
    const int rt = wave >> 1, chh = wave & 1;

    // zero local tables
    for (int i = tid; i < NC * ND; i += 256) ctab[i] = 0.0f;
    for (int i = tid; i < NC; i += 256) dtab[i] = 0.0f;
    // stage H1 weights as bf16
    const float4* wsrc = (const float4*)H1_w;
    for (int i = tid; i < ND * ND / 4; i += 256) {
        float4 v = wsrc[i];
        int b = i * 4;
        h1bf[b] = (__bf16)v.x; h1bf[b + 1] = (__bf16)v.y;
        h1bf[b + 2] = (__bf16)v.z; h1bf[b + 3] = (__bf16)v.w;
    }
    __syncthreads();

    for (int chunk = 0; chunk < 16; ++chunk) {
        const int row0 = blockIdx.x * 1024 + chunk * 64;
        // stage h tile (keep f32 for exact gating + ctx accumulation)
        const float4* hsrc = (const float4*)(h + (size_t)row0 * ND);
        for (int i = tid; i < 64 * ND / 4; i += 256) {
            float4 v = hsrc[i];
            int b = i * 4;
            ht[b] = v.x; ht[b + 1] = v.y; ht[b + 2] = v.z; ht[b + 3] = v.w;
        }
        if (tid < 64) {
            const int gi = row0 + tid;
            const int c = cid[gi];
            crow[tid] = c;
            const float m = dec_max(m_enc[c]);
            const float e = __expf(score[gi] - m);
            erow[tid] = e;
            atomicAdd(&dtab[c], e);            // ds_add_f32
        }
        __syncthreads();

        // ctx accumulation: ctab[cid[r]][d] += e_r * h[r][d]
        for (int i = tid; i < 64 * ND; i += 256) {
            const int r = i >> 7, d = i & (ND - 1);
            atomicAdd(&ctab[crow[r] * ND + d], erow[r] * ht[i]);  // ds_add_f32
        }

        // WMMA: a1 = h@H1^T for this 64-row chunk; gate & write h_merged
        v8f acc[4] = {v8f{}, v8f{}, v8f{}, v8f{}};
        const float* arow = ht + (rt * 16 + (lane & 15)) * ND;
#pragma unroll
        for (int kc = 0; kc < 4; ++kc) {
            const int k0 = kc * 32;
            v16bf a = frag_a_f32(arow, k0, lane);
#pragma unroll
            for (int c = 0; c < 4; ++c) {
                const int n = ((chh * 4 + c) * 16) + (lane & 15);
                v16bf b = frag_b_bf(h1bf + n * ND, k0, lane);
                acc[c] = __builtin_amdgcn_wmma_f32_16x16x32_bf16(
                    false, a, false, b, (short)0, acc[c], false, false);
            }
        }
        const int Mb = rt * 16 + ((lane >= 16) ? 8 : 0);
#pragma unroll
        for (int c = 0; c < 4; ++c) {
            const int col = (chh * 4 + c) * 16 + (lane & 15);
#pragma unroll
            for (int v = 0; v < 8; ++v) {
                const int R = Mb + v;
                const int cc = crow[R];
                const float a1 = acc[c][v] + u1[cc * ND + col];
                const float z  = sigmoidf_(a1);
                const float hv = ht[R * ND + col];
                const float gv = g_trans[cc * ND + col];
                out[(size_t)(row0 + R) * ND + col] = (1.0f - z) * hv + z * gv;
            }
        }
        __syncthreads();   // protect ht/erow/crow before next chunk
    }

    // flush per-block tables to global accumulators
    for (int i = tid; i < NC * ND; i += 256) {
        const float v = ctab[i];
        if (v != 0.0f) atomicAdd(&ctxe[i], v);
    }
    if (tid < NC) {
        const float v = dtab[tid];
        if (v != 0.0f) atomicAdd(&denom[tid], v);
    }
}

// ---------------------------------------------------------------------------
// Kernel 4: C-side finalize.  ctx = ctxe/denom; h_trans = tanh(ctx@W^T + W_b)
//   z2 = sigmoid(g@H2^T + H2_b + h_trans@G2^T + G2_b); g_merged -> out[B+c]
// ---------------------------------------------------------------------------
__global__ __launch_bounds__(ND) void k_cside_post(
    const float* __restrict__ g, const float* __restrict__ ctxe,
    const float* __restrict__ denom, const float* __restrict__ W_w,
    const float* __restrict__ W_b, const float* __restrict__ H2_w,
    const float* __restrict__ H2_b, const float* __restrict__ G2_w,
    const float* __restrict__ G2_b, float* __restrict__ out)
{
    __shared__ float ctxrow[ND], htr[ND], grow[ND];
    const int c = blockIdx.x, d = threadIdx.x;
    const float dn = fmaxf(denom[c], 1e-20f);
    ctxrow[d] = ctxe[c * ND + d] / dn;
    grow[d] = g[c * ND + d];
    __syncthreads();
    float s = W_b[d];
    const float* wr = W_w + d * ND;
#pragma unroll 4
    for (int k = 0; k < ND; ++k) s += ctxrow[k] * wr[k];
    htr[d] = tanhf(s);
    __syncthreads();
    float s2 = H2_b[d] + G2_b[d];
    const float* hr = H2_w + d * ND;
    const float* gr = G2_w + d * ND;
#pragma unroll 4
    for (int k = 0; k < ND; ++k) s2 += grow[k] * hr[k] + htr[k] * gr[k];
    const float z = sigmoidf_(s2);
    out[(size_t)(NB + c) * ND + d] = (1.0f - z) * grow[d] + z * htr[d];
}

// ---------------------------------------------------------------------------
extern "C" void kernel_launch(void* const* d_in, const int* in_sizes, int n_in,
                              void* d_out, int out_size, void* d_ws, size_t ws_size,
                              hipStream_t stream) {
    const float* h    = (const float*)d_in[0];
    const float* g    = (const float*)d_in[1];
    const int*   cid  = (const int*)d_in[2];
    const float* Wq   = (const float*)d_in[3];
    const float* Wk   = (const float*)d_in[4];
    const float* bias = (const float*)d_in[5];
    const float* Ws_w = (const float*)d_in[6];
    const float* Ws_b = (const float*)d_in[7];
    const float* W_w  = (const float*)d_in[8];
    const float* W_b  = (const float*)d_in[9];
    const float* Fs_w = (const float*)d_in[10];
    const float* Fs_b = (const float*)d_in[11];
    const float* H1_w = (const float*)d_in[12];
    const float* H1_b = (const float*)d_in[13];
    const float* G1_w = (const float*)d_in[14];
    const float* G1_b = (const float*)d_in[15];
    const float* H2_w = (const float*)d_in[16];
    const float* H2_b = (const float*)d_in[17];
    const float* G2_w = (const float*)d_in[18];
    const float* G2_b = (const float*)d_in[19];
    float* out = (float*)d_out;

    // workspace layout (floats): g_trans | q | u1 | score | m_enc | denom | ctxe
    char* ws = (char*)d_ws;
    float*    g_trans = (float*)ws;                   ws += NC * ND * 4;
    float*    qbuf    = (float*)ws;                   ws += NC * ND * 4;
    float*    u1      = (float*)ws;                   ws += NC * ND * 4;
    float*    score   = (float*)ws;                   ws += NB * 4;
    unsigned* m_enc   = (unsigned*)ws;                ws += NC * 4;
    float*    denom   = (float*)ws;                   ws += NC * 4;
    float*    ctxe    = (float*)ws;                   ws += NC * ND * 4;

    // zero the accumulators (m_enc, denom, ctxe are contiguous)
    hipMemsetAsync((void*)m_enc, 0, (size_t)(NC + NC + NC * ND) * 4, stream);

    k_cside_pre<<<NC, ND, 0, stream>>>(g, Fs_w, Fs_b, Wq, G1_w, G1_b, H1_b,
                                       g_trans, qbuf, u1);

    const size_t sh1 = (size_t)64 * ND * 2 + (size_t)ND * ND * 2 + (size_t)64 * ND * 4; // 80KB
    k_pass1<<<NB / 64, 256, sh1, stream>>>(h, Wk, cid, qbuf, bias, Ws_w, Ws_b,
                                           score, m_enc);

    const size_t sh2 = (size_t)NC * ND * 4 + (size_t)NC * 4 + (size_t)ND * ND * 2 +
                       (size_t)64 * ND * 4 + 64 * 4 + 64 * 4;                           // ~194KB
    k_pass2<<<NB / 1024, 256, sh2, stream>>>(h, cid, H1_w, u1, g_trans, score,
                                             m_enc, denom, ctxe, out);

    k_cside_post<<<NC, ND, 0, stream>>>(g, ctxe, denom, W_w, W_b, H2_w, H2_b,
                                        G2_w, G2_b, out);
}
